// VMM_10823317586413
// MI455X (gfx1250) — matrix operand order repaired and checked by reading, not compiled, not run on hardware
//
#include <hip/hip_runtime.h>
#include <hip/hip_bf16.h>
#include <math.h>

// ---------------- problem constants ----------------
#define Bb   8
#define Hh   64
#define Ww   64
#define DMc  96          // model dim
#define NSt  16          // state dim (DS)
#define Kk   4           // scan directions
#define DIc  192         // inner dim
#define Rr   6           // rank
#define Ll   4096        // H*W
#define RNr  38          // R + 2*N
#define MLr  (Bb*Ll)     // 32768 rows for token GEMMs
#define TILE 64          // scan LDS tile (L-steps per stage)

typedef __attribute__((ext_vector_type(16))) __bf16    v16bf;
typedef __attribute__((ext_vector_type(8)))  float     v8f;
typedef __attribute__((ext_vector_type(8)))  unsigned  v8u;
typedef __attribute__((ext_vector_type(4)))  unsigned  v4u;
typedef __attribute__((ext_vector_type(8)))  int       v8i;
typedef __attribute__((ext_vector_type(4)))  int       v4i;

// ---- TDM availability / arity probing ----
#if defined(__has_builtin)
#  if __has_builtin(__builtin_amdgcn_tensor_load_to_lds)
#    define HAVE_TDM 1
#  endif
#endif
#ifndef HAVE_TDM
#  define HAVE_TDM 0
#endif
#if defined(__has_include)
#  if __has_include(<hip/amd_detail/amd_gfx1250_TDM.h>)
#    define TDM_ARITY6 1
#  endif
#endif
#ifndef TDM_ARITY6
#  define TDM_ARITY6 0
#endif

// ---------------- helpers ----------------
// pack two f32 into two RNE bf16 in one dword
static __device__ __forceinline__ unsigned pack_bf16x2(float lo, float hi) {
  unsigned a = __builtin_bit_cast(unsigned, lo);
  unsigned b = __builtin_bit_cast(unsigned, hi);
  a += 0x7FFFu + ((a >> 16) & 1u);
  b += 0x7FFFu + ((b >> 16) & 1u);
  return (a >> 16) | (b & 0xFFFF0000u);
}
static __device__ __forceinline__ __bf16 to_bf16(float f) {
  unsigned u = __builtin_bit_cast(unsigned, f);
  u += 0x7FFFu + ((u >> 16) & 1u);
  unsigned short h = (unsigned short)(u >> 16);
  return __builtin_bit_cast(__bf16, h);
}
static __device__ __forceinline__ v16bf frag_zero() {
  v8u z = {0, 0, 0, 0, 0, 0, 0, 0};
  return __builtin_bit_cast(v16bf, z);
}
// A fragment (16x32 bf16, ISA 7.12.2) from packed bf16 row (p = &A[m][kk]):
// lane covers k = kh..kh+15 contiguous (kh = 16*(lane/16)) -> two b128 loads, no VALU.
static __device__ __forceinline__ v16bf load_a_frag_bf(const __bf16* p, int lane) {
  int kh = (lane & 16) ? 16 : 0;
  uint4 lo = *(const uint4*)(p + kh);
  uint4 hi = *(const uint4*)(p + kh + 8);
  v8u w = {lo.x, lo.y, lo.z, lo.w, hi.x, hi.y, hi.z, hi.w};
  return __builtin_bit_cast(v16bf, w);
}
// B fragment (32x16 bf16) from packed bf16 row (p = &Bt[n][kk]):
// lane covers k = khB..khB+15 contiguous -> two b128 loads.
static __device__ __forceinline__ v16bf load_b_frag_bf(const __bf16* p, int lane) {
  int kh = (lane & 16) ? 16 : 0;
  uint4 lo = *(const uint4*)(p + kh);
  uint4 hi = *(const uint4*)(p + kh + 8);
  v8u w = {lo.x, lo.y, lo.z, lo.w, hi.x, hi.y, hi.z, hi.w};
  return __builtin_bit_cast(v16bf, w);
}

static __device__ __forceinline__ float sigmoidf_(float x) { return 1.f / (1.f + __expf(-x)); }
static __device__ __forceinline__ float siluf_(float x)    { return x * sigmoidf_(x); }
static __device__ __forceinline__ float softplusf_(float x){ return (x > 20.f) ? x : log1pf(__expf(x)); }

#if HAVE_TDM
// Issue a TDM load of a RNr x TILE f32 tile (row stride Ll elements) into LDS.
// Descriptor per CDNA5 ISA: 2D tensor, data_size=4B.
static __device__ __forceinline__ void tdm_issue_tile(const float* g, void* lds) {
  unsigned long long ga = (unsigned long long)(size_t)g;
  unsigned la = (unsigned)(size_t)lds;     // flat LDS addr: low 32 bits = LDS offset
  v4u g0;
  g0.x = 1u;                                // count=1, user descriptor
  g0.y = la;                                // lds_addr
  g0.z = (unsigned)ga;                      // global_addr[31:0]
  g0.w = (unsigned)((ga >> 32) & 0x01FFFFFFull) | 0x80000000u; // addr[56:32] | type=2
  v8i g1;
  g1[0] = 0x00020000;                                     // data_size = 4 bytes
  g1[1] = (int)(((unsigned)Ll & 0xFFFFu) << 16);          // tensor_dim0 lo16 (bits 63:48)
  g1[2] = (int)(((unsigned)Ll >> 16) | ((unsigned)RNr << 16)); // dim0 hi | tensor_dim1 lo
  g1[3] = (int)((unsigned)TILE << 16);                    // tensor_dim1 hi=0 | tile_dim0
  g1[4] = RNr;                                            // tile_dim1 | tile_dim2=0
  g1[5] = Ll;                                             // tensor_dim0_stride lo32
  g1[6] = 0;
  g1[7] = 0;
  v4i gz = {0, 0, 0, 0};
#if TDM_ARITY6
  v8i gz8 = {0, 0, 0, 0, 0, 0, 0, 0};
  __builtin_amdgcn_tensor_load_to_lds(g0, g1, gz, gz, gz8, 0);
#else
  __builtin_amdgcn_tensor_load_to_lds(g0, g1, gz, gz, 0);
#endif
}
#endif

// ============================================================
// K0: one-shot f32 -> packed bf16 conversion (pairs -> dword stores)
// ============================================================
__global__ void k_f32_to_bf16(const float* __restrict__ src, unsigned* __restrict__ dst,
                              int n2) {
  int i = blockIdx.x * blockDim.x + threadIdx.x;
  if (i >= n2) return;
  float2 q = *(const float2*)(src + 2 * i);
  dst[i] = pack_bf16x2(q.x, q.y);
}

// ============================================================
// K1: xz = x @ in_proj_w^T ; split -> xi (b,DI,l) and z (b,l,DI)
// M=32768, N=384, K=96 -> 3 bf16 WMMA steps, direct bf16 fragment loads.
// ============================================================
__global__ void k_inproj(const __bf16* __restrict__ xbf, const __bf16* __restrict__ wbf,
                         float* __restrict__ xi, float* __restrict__ z) {
  int lane = threadIdx.x & 31, wave = threadIdx.x >> 5;
  int m0 = blockIdx.x * 16;
  int n0 = (blockIdx.y * 4 + wave) * 16;
  int lm = lane & 15;
  const __bf16* arow = xbf + (size_t)(m0 + lm) * DMc;
  const __bf16* brow = wbf + (size_t)(n0 + lm) * DMc;
  v8f acc = {};
  for (int kk = 0; kk < DMc; kk += 32) {
    if (kk + 32 < DMc) __builtin_prefetch(arow + kk + 32, 0, 0);
    v16bf a = load_a_frag_bf(arow + kk, lane);
    v16bf b = load_b_frag_bf(brow + kk, lane);
    acc = __builtin_amdgcn_wmma_f32_16x16x32_bf16(false, a, false, b,
                                                  (short)0, acc, false, false);
  }
  int bIdx = m0 / Ll;           // 16-row tile never crosses a batch
#pragma unroll
  for (int r = 0; r < 8; ++r) {
    int m = m0 + r + ((lane & 16) ? 8 : 0);
    int n = n0 + lm;
    int l = m & (Ll - 1);
    float v = acc[r];
    if (n < DIc) xi[((size_t)(bIdx * DIc + n)) * Ll + l] = v;
    else         z[(size_t)m * DIc + (n - DIc)] = v;
  }
}

// ============================================================
// K2: depthwise 3x3 conv (SAME) + bias + SiLU on (B,DI,H,W)
// ============================================================
__global__ void k_conv(const float* __restrict__ xi, const float* __restrict__ cw,
                       const float* __restrict__ cb, float* __restrict__ xc) {
  int idx = blockIdx.x * blockDim.x + threadIdx.x;
  if (idx >= Bb * DIc * Ll) return;
  int l = idx & (Ll - 1);
  int bd = idx >> 12;
  int d = bd % DIc;
  int h = l >> 6, w = l & 63;
  const float* src = xi + (size_t)bd * Ll;
  float acc = cb[d];
#pragma unroll
  for (int ky = 0; ky < 3; ++ky) {
    int hy = h + ky - 1;
    if ((unsigned)hy >= (unsigned)Hh) continue;
#pragma unroll
    for (int kx = 0; kx < 3; ++kx) {
      int wx = w + kx - 1;
      if ((unsigned)wx >= (unsigned)Ww) continue;
      acc += cw[d * 9 + ky * 3 + kx] * src[hy * Ww + wx];
    }
  }
  xc[idx] = siluf_(acc);
}

// ============================================================
// K3: x_dbl[b,k,rn,l] = sum_d xpw[k,rn,d] * xc[b,d,scan_ids[k,l]]
// A = bf16 weights (direct loads), B = gathered f32 columns (pack in-flight).
// ============================================================
__global__ void k_xdbl(const __bf16* __restrict__ xpwbf, const float* __restrict__ xc,
                       const int* __restrict__ mair, float* __restrict__ xdbl) {
  int lane = threadIdx.x & 31, wave = threadIdx.x >> 5;
  int l0 = (blockIdx.x * 4 + wave) * 16;
  int r0 = blockIdx.y * 16;
  int bk = blockIdx.z;
  int b = bk >> 2, k = bk & 3;
  int lm = lane & 15;
  int khB = (lane & 16) ? 16 : 0;
  int sid = mair[k * Ll + l0 + lm];
  const float* xcb = xc + (size_t)b * DIc * Ll;
  int rn = r0 + lm;
  bool aval = rn < RNr;
  const __bf16* apw = xpwbf + (size_t)(k * RNr + (aval ? rn : 0)) * DIc;
  v8f acc = {};
  for (int kk = 0; kk < DIc; kk += 32) {
    v16bf a = aval ? load_a_frag_bf(apw + kk, lane) : frag_zero();
    v8u wv;
#pragma unroll
    for (int j = 0; j < 8; ++j) {
      int kd = kk + khB + 2 * j;
      wv[j] = pack_bf16x2(xcb[(size_t)kd * Ll + sid],
                          xcb[(size_t)(kd + 1) * Ll + sid]);
    }
    v16bf bfr = __builtin_bit_cast(v16bf, wv);
    acc = __builtin_amdgcn_wmma_f32_16x16x32_bf16(false, a, false, bfr,
                                                  (short)0, acc, false, false);
  }
#pragma unroll
  for (int r = 0; r < 8; ++r) {
    int rr = r0 + r + ((lane & 16) ? 8 : 0);
    if (rr < RNr)
      xdbl[((size_t)bk * RNr + rr) * Ll + l0 + lm] = acc[r];
  }
}

// ============================================================
// K4: selective scan. Block per (b,k); thread = channel d; 16 states in VGPRs.
// dt_proj + softplus + Ds*u skip fused. Tile staging of the 38xTILE x_dbl slab
// is done by the Tensor Data Mover, double-buffered against compute.
// ============================================================
__global__ void k_scan(const float* __restrict__ xdbl, const float* __restrict__ xc,
                       const int* __restrict__ mair,
                       const float* __restrict__ dtw, const float* __restrict__ dtb,
                       const float* __restrict__ alogs, const float* __restrict__ Ds,
                       float* __restrict__ outy) {
  __shared__ float s_tile[2][RNr][TILE];
  __shared__ int   s_id[2][TILE];
  int bk = blockIdx.x;
  int b = bk >> 2, k = bk & 3;
  int d = threadIdx.x;
  int kd = k * DIc + d;
  float An[NSt];
#pragma unroll
  for (int n = 0; n < NSt; ++n) An[n] = -__expf(alogs[kd * NSt + n]);
  float wdt[Rr];
#pragma unroll
  for (int r = 0; r < Rr; ++r) wdt[r] = dtw[kd * Rr + r];
  float bias = dtb[kd];
  float Dsv = Ds[kd];
  float s[NSt];
#pragma unroll
  for (int n = 0; n < NSt; ++n) s[n] = 0.f;

  const float* xrow = xdbl + (size_t)bk * RNr * Ll;
  const float* xcb  = xc + ((size_t)b * DIc + d) * Ll;
  float* orow = outy + ((size_t)bk * DIc + d) * Ll;
  const int* sidk = mair + k * Ll;
  const int NT = Ll / TILE;

  // ---- prologue: stage tile 0 ----
#if HAVE_TDM
  if (threadIdx.x == 0) tdm_issue_tile(xrow, &s_tile[0][0][0]);
#else
  for (int idx = threadIdx.x; idx < TILE * RNr; idx += blockDim.x)
    s_tile[0][idx / TILE][idx % TILE] = xrow[(size_t)(idx / TILE) * Ll + (idx % TILE)];
#endif
  for (int idx = threadIdx.x; idx < TILE; idx += blockDim.x) s_id[0][idx] = sidk[idx];

  for (int it = 0; it < NT; ++it) {
    int cur = it & 1;
    int l0 = it * TILE;
#if HAVE_TDM
    if (threadIdx.x < 32) __builtin_amdgcn_s_wait_tensorcnt(0);  // wave 0 owns TENSORcnt
#endif
    __syncthreads();
    if (it + 1 < NT) {                       // stage next tile into the other buffer
      int nxt = cur ^ 1;
      int ln = l0 + TILE;
#if HAVE_TDM
      if (threadIdx.x == 0) tdm_issue_tile(xrow + ln, &s_tile[nxt][0][0]);
#else
      for (int idx = threadIdx.x; idx < TILE * RNr; idx += blockDim.x)
        s_tile[nxt][idx / TILE][idx % TILE] =
            xrow[(size_t)(idx / TILE) * Ll + ln + (idx % TILE)];
#endif
      for (int idx = threadIdx.x; idx < TILE; idx += blockDim.x)
        s_id[nxt][idx] = sidk[ln + idx];
    }
    // ---- 64 recurrence steps from LDS (all reads are broadcasts) ----
    for (int t = 0; t < TILE; ++t) {
      float u = xcb[s_id[cur][t]];
      float dv = bias;
#pragma unroll
      for (int r = 0; r < Rr; ++r) dv += wdt[r] * s_tile[cur][r][t];
      float dl = softplusf_(dv);
      float du = dl * u;
      float y = 0.f;
#pragma unroll
      for (int n = 0; n < NSt; ++n) {
        float sn = s[n] * __expf(dl * An[n]) + du * s_tile[cur][Rr + n][t];
        s[n] = sn;
        y += sn * s_tile[cur][Rr + NSt + n][t];
      }
      orow[l0 + t] = y + Dsv * u;
    }
  }
}

// ============================================================
// K5: p[b,d,k] = mean_l outy[b,k,d,l] (mean invariant under inverse gather)
// ============================================================
__global__ void k_pool(const float* __restrict__ outy, float* __restrict__ p) {
  int gid = blockIdx.x;                 // = bk*DI + d
  int d = gid % DIc;
  int bk = gid / DIc;
  int b = bk >> 2, k = bk & 3;
  const float* row = outy + (size_t)gid * Ll;
  float sum = 0.f;
  for (int i = threadIdx.x; i < Ll; i += blockDim.x) sum += row[i];
  __shared__ float sh[256];
  sh[threadIdx.x] = sum;
  __syncthreads();
  for (int st = 128; st > 0; st >>= 1) {
    if (threadIdx.x < st) sh[threadIdx.x] += sh[threadIdx.x + st];
    __syncthreads();
  }
  if (threadIdx.x == 0) p[(b * DIc + d) * Kk + k] = sh[0] * (1.f / Ll);
}

// ============================================================
// K6: scale[b,k,d] = sigmoid( sum_i p[b,d,i]*gw[d,k,i] + gb[d,k] )
// ============================================================
__global__ void k_gate(const float* __restrict__ p, const float* __restrict__ gw,
                       const float* __restrict__ gb, float* __restrict__ scale) {
  int idx = blockIdx.x * blockDim.x + threadIdx.x;
  if (idx >= Bb * Kk * DIc) return;
  int d = idx % DIc;
  int bkk = idx / DIc;
  int b = bkk >> 2, k = bkk & 3;
  float g = gb[d * Kk + k];
#pragma unroll
  for (int i = 0; i < Kk; ++i)
    g += p[(b * DIc + d) * Kk + i] * gw[d * Kk * Kk + k * Kk + i];
  scale[idx] = sigmoidf_(g);
}

// ============================================================
// K7: inverse gather + weighted K-sum + LayerNorm(DI) + *silu(z)
// emits yln directly as packed bf16 (it feeds a bf16 WMMA GEMM)
// ============================================================
__global__ void k_combine(const float* __restrict__ outy, const int* __restrict__ mair,
                          const float* __restrict__ scale, const float* __restrict__ onw,
                          const float* __restrict__ onb, const float* __restrict__ z,
                          __bf16* __restrict__ ylnbf) {
  __shared__ float sh[256];
  int m = blockIdx.x;
  int b = m >> 12, l = m & (Ll - 1);
  int d = threadIdx.x;
  float v = 0.f;
#pragma unroll
  for (int k = 0; k < Kk; ++k) {
    int li = mair[(Kk + k) * Ll + l];   // inv_ids block follows scan_ids
    v += scale[(b * Kk + k) * DIc + d] *
         outy[((size_t)(b * Kk + k) * DIc + d) * Ll + li];
  }
  if (threadIdx.x < 64) sh[192 + threadIdx.x] = 0.f;
  sh[d] = v;
  __syncthreads();
  for (int st = 128; st > 0; st >>= 1) {
    if (d < st) sh[d] += sh[d + st];
    __syncthreads();
  }
  float mu = sh[0] * (1.f / DIc);
  __syncthreads();
  float dv = v - mu;
  if (threadIdx.x < 64) sh[192 + threadIdx.x] = 0.f;
  sh[d] = dv * dv;
  __syncthreads();
  for (int st = 128; st > 0; st >>= 1) {
    if (d < st) sh[d] += sh[d + st];
    __syncthreads();
  }
  float rstd = rsqrtf(sh[0] * (1.f / DIc) + 1e-5f);
  float zz = z[(size_t)m * DIc + d];
  float r = (dv * rstd * onw[d] + onb[d]) * siluf_(zz);
  ylnbf[(size_t)m * DIc + d] = to_bf16(r);
}

// ============================================================
// K8: out = yln @ out_proj_w^T ; M=32768, N=96, K=192 -> 6 WMMA steps
// pure bf16 fragment loads, f32 accumulate/store.
// ============================================================
__global__ void k_outproj(const __bf16* __restrict__ ylnbf, const __bf16* __restrict__ opwbf,
                          float* __restrict__ out) {
  int lane = threadIdx.x & 31, wave = threadIdx.x >> 5;
  int m0 = blockIdx.x * 16;
  int n0 = (blockIdx.y * 3 + wave) * 16;
  int lm = lane & 15;
  const __bf16* arow = ylnbf + (size_t)(m0 + lm) * DIc;
  const __bf16* brow = opwbf + (size_t)(n0 + lm) * DIc;
  v8f acc = {};
  for (int kk = 0; kk < DIc; kk += 32) {
    if (kk + 32 < DIc) __builtin_prefetch(arow + kk + 32, 0, 0);
    v16bf a = load_a_frag_bf(arow + kk, lane);
    v16bf b = load_b_frag_bf(brow + kk, lane);
    acc = __builtin_amdgcn_wmma_f32_16x16x32_bf16(false, a, false, b,
                                                  (short)0, acc, false, false);
  }
#pragma unroll
  for (int r = 0; r < 8; ++r) {
    int m = m0 + r + ((lane & 16) ? 8 : 0);
    out[(size_t)m * DMc + n0 + lm] = acc[r];
  }
}

// ---------------- host-side launcher ----------------
extern "C" void kernel_launch(void* const* d_in, const int* in_sizes, int n_in,
                              void* d_out, int out_size, void* d_ws, size_t ws_size,
                              hipStream_t stream) {
  const float* x      = (const float*)d_in[0];
  const int*   mair   = (const int*)d_in[1];
  const float* ipw    = (const float*)d_in[2];
  const float* cw     = (const float*)d_in[3];
  const float* cb     = (const float*)d_in[4];
  const float* xpw    = (const float*)d_in[5];
  const float* dtw    = (const float*)d_in[6];
  const float* dtb    = (const float*)d_in[7];
  const float* alogs  = (const float*)d_in[8];
  const float* Ds     = (const float*)d_in[9];
  const float* onw    = (const float*)d_in[10];
  const float* onb    = (const float*)d_in[11];
  const float* opw    = (const float*)d_in[12];
  const float* gw     = (const float*)d_in[13];
  const float* gb     = (const float*)d_in[14];
  float* out = (float*)d_out;

  float* ws = (float*)d_ws;
  size_t off = 0;
  float* xi    = ws + off; off += (size_t)Bb * DIc * Ll;
  float* zbuf  = ws + off; off += (size_t)Bb * Ll * DIc;
  float* xc    = ws + off; off += (size_t)Bb * DIc * Ll;
  float* xdbl  = ws + off; off += (size_t)Bb * Kk * RNr * Ll;
  float* outy  = ws + off; off += (size_t)Bb * Kk * DIc * Ll;
  float* pbuf  = ws + off; off += (size_t)Bb * DIc * Kk;
  float* scale = ws + off; off += (size_t)Bb * Kk * DIc;
  // packed bf16 regions (allocated in float units, all 16B aligned)
  __bf16* xbf   = (__bf16*)(ws + off); off += (size_t)MLr * DMc / 2;
  __bf16* ipwbf = (__bf16*)(ws + off); off += (size_t)2 * DIc * DMc / 2;
  __bf16* xpwbf = (__bf16*)(ws + off); off += (size_t)Kk * RNr * DIc / 2;
  __bf16* opwbf = (__bf16*)(ws + off); off += (size_t)DMc * DIc / 2;
  __bf16* ylnbf = (__bf16*)(ws + off); off += (size_t)MLr * DIc / 2;

  // one-shot bf16 conversions of GEMM operands
  {
    int n2 = MLr * DMc / 2;
    k_f32_to_bf16<<<(n2 + 255) / 256, 256, 0, stream>>>(x, (unsigned*)xbf, n2);
    n2 = 2 * DIc * DMc / 2;
    k_f32_to_bf16<<<(n2 + 255) / 256, 256, 0, stream>>>(ipw, (unsigned*)ipwbf, n2);
    n2 = Kk * RNr * DIc / 2;
    k_f32_to_bf16<<<(n2 + 255) / 256, 256, 0, stream>>>(xpw, (unsigned*)xpwbf, n2);
    n2 = DMc * DIc / 2;
    k_f32_to_bf16<<<(n2 + 255) / 256, 256, 0, stream>>>(opw, (unsigned*)opwbf, n2);
  }

  k_inproj<<<dim3(MLr / 16, 6), 128, 0, stream>>>(xbf, ipwbf, xi, zbuf);
  k_conv<<<(Bb * DIc * Ll + 255) / 256, 256, 0, stream>>>(xi, cw, cb, xc);
  k_xdbl<<<dim3(Ll / 64, 3, Bb * Kk), 128, 0, stream>>>(xpwbf, xc, mair, xdbl);
  k_scan<<<Bb * Kk, DIc, 0, stream>>>(xdbl, xc, mair, dtw, dtb, alogs, Ds, outy);
  k_pool<<<Bb * Kk * DIc, 256, 0, stream>>>(outy, pbuf);
  k_gate<<<(Bb * Kk * DIc + 255) / 256, 256, 0, stream>>>(pbuf, gw, gb, scale);
  k_combine<<<MLr, DIc, 0, stream>>>(outy, mair, scale, onw, onb, zbuf, ylnbf);
  k_outproj<<<dim3(MLr / 16, 2), 96, 0, stream>>>(ylnbf, opwbf, out);
}
